// PositionActor_38886633898255
// MI455X (gfx1250) — matrix operand order repaired and checked by reading, not compiled, not run on hardware
//
#include <hip/hip_runtime.h>
#include <cmath>

// ---------------------------------------------------------------------------
// MI455X / gfx1250 fused PositionActor kernel.
//   scores[b,p] = relu([seq[b,p], seq[b,p+1]] @ W1 + b1) @ W2 + b2
// Compute-bound GEMM (2.75e11 FLOP) on v_wmma_f32_16x16x32_bf16; seq staged
// once per tile into 258 KB of the 320 KB WGP LDS; W1 slices double-buffered;
// b1/W2 fetched via global_load_async_to_lds (ASYNCcnt).
// ---------------------------------------------------------------------------

typedef __bf16 bf16;
typedef bf16  v16bf __attribute__((ext_vector_type(16)));
typedef bf16  v8bf  __attribute__((ext_vector_type(8)));
typedef float v8f   __attribute__((ext_vector_type(8)));
typedef float v4f   __attribute__((ext_vector_type(4)));

#define DIM    512            // EMB
#define KDIM   1024           // 2*EMB (GEMM K)
#define NDIM   1024           // hidden width (GEMM N)
#define TM     128            // pair rows per block
#define TN     128            // hidden cols per chunk
#define KSTEP  64             // K per double-buffered stage (2 WMMA k-halves)
#define NK     (KDIM / KSTEP) // 16 stages
#define AROW   (KDIM + 8)     // bf16 row stride for A tile (2064B: 16B align, bank-skewed)
#define BROW   (KSTEP + 8)    // bf16 row stride for W1 slice^T (144B: 16B align, skewed)

#define AS_ELEMS  ((size_t)TM * AROW)
#define BS_ELEMS  ((size_t)TN * BROW)

__device__ __forceinline__ void stage_load(const float* __restrict__ src0,
                                           v4f ta[4], v4f tb[4])
{
    #pragma unroll
    for (int j = 0; j < 4; ++j) {
        ta[j] = *(const v4f*)(src0 + 4 * j);          // row k0+kk2
        tb[j] = *(const v4f*)(src0 + NDIM + 4 * j);   // row k0+kk2+1
    }
}

__device__ __forceinline__ void stage_store(bf16* __restrict__ dst, int nnB, int kk2,
                                            const v4f ta[4], const v4f tb[4])
{
    #pragma unroll
    for (int j = 0; j < 4; ++j)
        #pragma unroll
        for (int i = 0; i < 4; ++i) {
            union { bf16 h[2]; unsigned u; } pk;
            pk.h[0] = (bf16)ta[j][i];
            pk.h[1] = (bf16)tb[j][i];
            *(unsigned*)(dst + (size_t)(nnB + 4 * j + i) * BROW + kk2) = pk.u;
        }
}

__global__ __launch_bounds__(256)
void pair_mlp_scores(const float* __restrict__ seq,   // [B, L, 512]
                     const float* __restrict__ W1,    // [1024, 1024] (e,f) row-major
                     const float* __restrict__ b1,    // [1024]
                     const float* __restrict__ W2,    // [1024]
                     const float* __restrict__ b2,    // [1]
                     float* __restrict__ scores,      // [B, L-1]
                     int B, int L)
{
    extern __shared__ char smem[];
    bf16*  As    = (bf16*)smem;                                   // [TM][AROW]
    bf16*  Bbuf  = (bf16*)(smem + AS_ELEMS * 2);                  // 2 x [TN][BROW]
    float* Lb1   = (float*)(smem + AS_ELEMS * 2 + 2 * BS_ELEMS * 2);
    float* Lw2   = Lb1 + NDIM;
    float* sPart = Lw2 + NDIM;

    const int tid  = threadIdx.x;
    const int wave = tid >> 5;
    const int lane = tid & 31;
    const int half = lane >> 4;
    const int l16  = lane & 15;

    const int nP      = L - 1;
    const int tilesPB = (nP + TM - 1) / TM;
    const int b       = blockIdx.x / tilesPB;
    const int pBase   = (blockIdx.x % tilesPB) * TM;
    const float bias2 = b2[0];

    if (tid < TM) sPart[tid] = 0.0f;

    // ---- b1 / W2 -> LDS via CDNA5 async DMA (overlaps the A staging below) --
    {
        unsigned o1 = (unsigned)(unsigned long long)(void*)(Lb1 + tid * 4);
        asm volatile("global_load_async_to_lds_b128 %0, %1, off"
                     :: "v"(o1), "v"(b1 + tid * 4) : "memory");
        unsigned o2 = (unsigned)(unsigned long long)(void*)(Lw2 + tid * 4);
        asm volatile("global_load_async_to_lds_b128 %0, %1, off"
                     :: "v"(o2), "v"(W2 + tid * 4) : "memory");
    }

    // ---- Stage A tile once: 128 pair-rows x K=1024, fp32 -> bf16 in LDS ----
    {
        const int row = tid >> 1;          // 0..127
        const int hh  = tid & 1;           // 0: k<512 (seq row p), 1: k>=512 (row p+1)
        int srow = pBase + row + hh;
        if (srow > L - 1) srow = L - 1;
        const float* src = seq + ((size_t)b * L + srow) * DIM;
        bf16* dst = As + (size_t)row * AROW + hh * DIM;
        #pragma unroll 4
        for (int j = 0; j < DIM; j += 8) {
            v4f a = *(const v4f*)(src + j);
            v4f c = *(const v4f*)(src + j + 4);
            v8bf o;
            o[0] = (bf16)a[0]; o[1] = (bf16)a[1]; o[2] = (bf16)a[2]; o[3] = (bf16)a[3];
            o[4] = (bf16)c[0]; o[5] = (bf16)c[1]; o[6] = (bf16)c[2]; o[7] = (bf16)c[3];
            *(v8bf*)(dst + j) = o;
        }
    }
    asm volatile("s_wait_asynccnt 0" ::: "memory");
    __syncthreads();

    const int kk2 = (tid & 31) * 2;        // k-pair this thread stages
    const int nnB = (tid >> 5) * 16;       // 16 n-columns this thread stages
    const bf16* apRow = As + (size_t)(wave * 16 + l16) * AROW + half * 8;

    for (int n0 = 0; n0 < NDIM; n0 += TN) {
        v8f acc[8];
        #pragma unroll
        for (int t = 0; t < 8; ++t)
            #pragma unroll
            for (int i = 0; i < 8; ++i) acc[t][i] = 0.0f;

        // Prime stage 0 into buffer 0.
        {
            v4f ta[4], tb[4];
            const float* src0 = W1 + (size_t)kk2 * NDIM + n0 + nnB;
            stage_load(src0, ta, tb);
            stage_store(Bbuf, nnB, kk2, ta, tb);
        }
        __syncthreads();

        for (int ks = 0; ks < NK; ++ks) {
            bf16* cur = Bbuf + (size_t)(ks & 1) * BS_ELEMS;
            bf16* nxt = Bbuf + (size_t)((ks & 1) ^ 1) * BS_ELEMS;
            const bool more = (ks + 1 < NK);

            // 1) issue global loads for the NEXT W1 slice (latency hides under WMMA)
            v4f ta[4], tb[4];
            if (more) {
                const float* src0 =
                    W1 + (size_t)((ks + 1) * KSTEP + kk2) * NDIM + n0 + nnB;
                stage_load(src0, ta, tb);
                if (ks + 2 < NK)
                    __builtin_prefetch(src0 + (size_t)KSTEP * NDIM, 0, 1);
            }

            // 2) compute on the CURRENT slice: 2 k-halves x 8 WMMA
            #pragma unroll
            for (int kh = 0; kh < 2; ++kh) {
                const int kb = ks * KSTEP + kh * 32;
                // A fragment: lane half 0 -> K {0-7,16-23}, half 1 -> K {8-15,24-31}
                union { v16bf v; v8bf h[2]; } ua;
                const bf16* ap = apRow + kb;
                ua.h[0] = *(const v8bf*)(ap);
                ua.h[1] = *(const v8bf*)(ap + 16);
                // Preload all 8 B fragments, then run the WMMA chain back-to-back.
                v16bf bfr[8];
                #pragma unroll
                for (int t = 0; t < 8; ++t) {
                    const bf16* bp =
                        cur + (size_t)(t * 16 + l16) * BROW + kh * 32 + half * 8;
                    union { v16bf v; v8bf h[2]; } ub;
                    ub.h[0] = *(const v8bf*)(bp);
                    ub.h[1] = *(const v8bf*)(bp + 16);
                    bfr[t] = ub.v;
                }
                #pragma unroll
                for (int t = 0; t < 8; ++t)
                    acc[t] = __builtin_amdgcn_wmma_f32_16x16x32_bf16(
                        false, ua.v, false, bfr[t], (short)0, acc[t], false, false);
            }

            // 3) commit the staged slice into the other buffer
            if (more) stage_store(nxt, nnB, kk2, ta, tb);
            __syncthreads();
        }

        // Epilogue: h = relu(acc + b1), fold the W2 reduction into row partials.
        // C/D layout: VGPR v, lane half h -> row M = v + 8h; col N = l16 (tile t).
        float rsum[8];
        #pragma unroll
        for (int v = 0; v < 8; ++v) rsum[v] = 0.0f;
        #pragma unroll
        for (int t = 0; t < 8; ++t) {
            const int col = n0 + t * 16 + l16;
            const float bb = Lb1[col];
            const float ww = Lw2[col];
            #pragma unroll
            for (int v = 0; v < 8; ++v) {
                float h = acc[t][v] + bb;
                if (h > 0.0f) rsum[v] += h * ww;
            }
        }
        #pragma unroll
        for (int v = 0; v < 8; ++v)
            atomicAdd(&sPart[wave * 16 + v + half * 8], rsum[v]);   // ds_add_f32
        __syncthreads();
    }

    if (tid < TM) {
        const int p = pBase + tid;
        if (p < nP) scores[(size_t)b * nP + p] = sPart[tid] + bias2;
    }
}

// ---------------------------------------------------------------------------
// Per-batch masked softmax stats: argmax action (first occurrence),
// logprob(argmax) = -log Z, entropy = log Z - (sum e_i * d_i)/Z.
// ---------------------------------------------------------------------------
__global__ __launch_bounds__(256)
void softmax_stats(const float* __restrict__ scores,
                   const int* __restrict__ lengths,
                   void* __restrict__ out,
                   int B, int L, int out_mode)
{
    const int b   = blockIdx.x;
    const int tid = threadIdx.x;
    const int nP  = L - 1;
    int nValid = lengths[b] - 1;
    if (nValid < 1) nValid = 1;
    if (nValid > nP) nValid = nP;
    const float* s = scores + (size_t)b * nP;

    __shared__ float smax[256];
    __shared__ int   sidx[256];
    __shared__ float ssum[256];
    __shared__ float ssum2[256];

    float m = -INFINITY; int mi = 0x7fffffff;
    for (int p = tid; p < nValid; p += 256) {
        float v = s[p];
        if (v > m) { m = v; mi = p; }
    }
    smax[tid] = m; sidx[tid] = mi;
    __syncthreads();
    for (int st = 128; st > 0; st >>= 1) {
        if (tid < st) {
            float om = smax[tid + st]; int oi = sidx[tid + st];
            if (om > smax[tid] || (om == smax[tid] && oi < sidx[tid])) {
                smax[tid] = om; sidx[tid] = oi;
            }
        }
        __syncthreads();
    }
    m = smax[0]; mi = sidx[0];

    float z = 0.0f, e2 = 0.0f;
    for (int p = tid; p < nValid; p += 256) {
        float d  = s[p] - m;
        float ex = expf(d);
        z  += ex;
        e2 += ex * d;
    }
    ssum[tid] = z; ssum2[tid] = e2;
    __syncthreads();
    for (int st = 128; st > 0; st >>= 1) {
        if (tid < st) { ssum[tid] += ssum[tid + st]; ssum2[tid] += ssum2[tid + st]; }
        __syncthreads();
    }
    if (tid == 0) {
        float Z = ssum[0], E = ssum2[0];
        float logZ    = logf(Z);
        float logprob = -logZ;              // score at argmax equals the max
        float entropy = logZ - E / Z;
        if (out_mode == 1) {                // [i64 action xB][f32 logprob xB][f32 entropy xB]
            ((long long*)out)[b] = (long long)mi;
            float* f = (float*)out;
            f[2 * B + b] = logprob;
            f[3 * B + b] = entropy;
        } else {                            // all-float flattening
            float* f = (float*)out;
            f[b]         = (float)mi;
            f[B + b]     = logprob;
            f[2 * B + b] = entropy;
        }
    }
}

extern "C" void kernel_launch(void* const* d_in, const int* in_sizes, int n_in,
                              void* d_out, int out_size, void* d_ws, size_t ws_size,
                              hipStream_t stream)
{
    const float* seq     = (const float*)d_in[0];   // (B, L, 512) f32
    const float* W1      = (const float*)d_in[1];   // (1024, 1024) f32
    const float* b1      = (const float*)d_in[2];   // (1024,)
    const float* W2      = (const float*)d_in[3];   // (1024, 1)
    const float* b2      = (const float*)d_in[4];   // (1,)
    const int*   lengths = (const int*)d_in[5];     // (B,) i32

    const int B  = in_sizes[5];
    const int L  = in_sizes[0] / (B * DIM);
    const int nP = L - 1;

    float* scores = (float*)d_ws;                   // B * (L-1) f32 (~524 KB)

    const int tilesPB = (nP + TM - 1) / TM;
    const size_t shmem = AS_ELEMS * sizeof(bf16)          // A tile (~258 KB)
                       + 2 * BS_ELEMS * sizeof(bf16)      // double-buffered W1 slice^T
                       + 2 * NDIM * sizeof(float)         // b1 + W2 (async DMA'd)
                       + TM * sizeof(float);              // score partials

    // CDNA5 WGP has 320 KB LDS; raise the dynamic-LDS cap for this kernel.
    (void)hipFuncSetAttribute((const void*)pair_mlp_scores,
                              hipFuncAttributeMaxDynamicSharedMemorySize,
                              (int)shmem);

    pair_mlp_scores<<<B * tilesPB, 256, shmem, stream>>>(
        seq, W1, b1, W2, b2, scores, B, L);

    const int out_mode = (out_size == 4 * B) ? 1 : 0;
    softmax_stats<<<B, 256, 0, stream>>>(scores, lengths, d_out, B, L, out_mode);
}
